// MSDeformAttnTransformerEncoderLayer_21689584844815
// MI455X (gfx1250) — matrix operand order repaired and checked by reading, not compiled, not run on hardware
//
#include <hip/hip_runtime.h>

// ---------------------------------------------------------------------------
// MSDeformAttn transformer encoder layer for gfx1250 (MI455X).
//   B=2, LEN=21760, D_MODEL=256, D_FFN=1024, HEADS=8, LEVELS=4, POINTS=4,
//   HEAD_DIM=32 (== wave32 width).
// All dense GEMMs run through v_wmma_f32_16x16x32_f16 with f32 accumulate.
// ---------------------------------------------------------------------------

#define LEN_Q   21760
#define BATCH   2
#define ROWS    (BATCH * LEN_Q)     // 43520, multiple of 128
#define DMODEL  256
#define DFFN    1024

typedef __attribute__((ext_vector_type(16))) _Float16 v16h;
typedef __attribute__((ext_vector_type(8)))  _Float16 v8h;
typedef __attribute__((ext_vector_type(8)))  float    v8f;

// ---------------------------------------------------------------------------
// Generic WMMA GEMM:  C[M,N] = A[M,K](f16) * Bt[N,K](f16)^T + bias
// Optional residual add, exact GELU, f32 and/or f16 outputs.
// Block = 256 threads = 8 waves; wave computes a 16(M) x 64(N) tile.
// Grid: x = M/128, y = N/64.
// ---------------------------------------------------------------------------
__global__ __launch_bounds__(256)
void wmma_gemm_kernel(const _Float16* __restrict__ A,
                      const _Float16* __restrict__ Bt,
                      const float*    __restrict__ bias,
                      const float*    __restrict__ resid,
                      float*          __restrict__ Cf32,
                      _Float16*       __restrict__ Cf16,
                      int M, int K, int N, int do_gelu)
{
    const int wave = threadIdx.x >> 5;
    const int lane = threadIdx.x & 31;
    const int row  = lane & 15;        // C column index within tile / A row
    const int hi   = lane >> 4;        // half-wave selector

    const int m0 = blockIdx.x * 128 + wave * 16;
    const int n0 = blockIdx.y * 64;
    if (m0 >= M) return;

    const _Float16* arow = A + (size_t)(m0 + row) * K;
    const _Float16* brow0 = Bt + (size_t)(n0 +  0 + row) * K;
    const _Float16* brow1 = Bt + (size_t)(n0 + 16 + row) * K;
    const _Float16* brow2 = Bt + (size_t)(n0 + 32 + row) * K;
    const _Float16* brow3 = Bt + (size_t)(n0 + 48 + row) * K;

    v8f acc[4] = {};

    for (int k0 = 0; k0 < K; k0 += 32) {
        // A fragment (16x32 f16): lanes 0-15 -> M rows, halves 0..15 = K hi*16 ..
        // hi*16+15 contiguous -> two contiguous 16B loads per lane.
        union { v16h v; v8h h[2]; } af;
        af.h[0] = *(const v8h*)(arow + k0 + hi * 16);
        af.h[1] = *(const v8h*)(arow + k0 + hi * 16 + 8);

        // B fragment (32x16 f16): lane -> column, halves = K hi*16 .. hi*16+15
        // contiguous -> one 32B load per lane (pre-transposed weights [N,K]).
        v16h bf0 = *(const v16h*)(brow0 + k0 + hi * 16);
        v16h bf1 = *(const v16h*)(brow1 + k0 + hi * 16);
        v16h bf2 = *(const v16h*)(brow2 + k0 + hi * 16);
        v16h bf3 = *(const v16h*)(brow3 + k0 + hi * 16);

        if (k0 + 32 < K) {
            __builtin_prefetch(arow  + k0 + 32, 0, 3);   // global_prefetch_b8
            __builtin_prefetch(brow0 + k0 + 32, 0, 3);
        }

        acc[0] = __builtin_amdgcn_wmma_f32_16x16x32_f16(false, af.v, false, bf0,
                                                        (short)0, acc[0], false, false);
        acc[1] = __builtin_amdgcn_wmma_f32_16x16x32_f16(false, af.v, false, bf1,
                                                        (short)0, acc[1], false, false);
        acc[2] = __builtin_amdgcn_wmma_f32_16x16x32_f16(false, af.v, false, bf2,
                                                        (short)0, acc[2], false, false);
        acc[3] = __builtin_amdgcn_wmma_f32_16x16x32_f16(false, af.v, false, bf3,
                                                        (short)0, acc[3], false, false);
    }

    // Epilogue. C layout: lane -> column (row), VGPR r -> M = hi*8 + r.
    #pragma unroll
    for (int j = 0; j < 4; ++j) {
        const int n = n0 + j * 16 + row;
        const float bv = bias[n];
        #pragma unroll
        for (int r = 0; r < 8; ++r) {
            const int m = m0 + hi * 8 + r;
            float v = acc[j][r] + bv;
            if (resid)  v += resid[(size_t)m * N + n];
            if (do_gelu) v = 0.5f * v * (1.0f + erff(v * 0.70710678118654752f));
            if (Cf32) Cf32[(size_t)m * N + n] = v;
            if (Cf16) Cf16[(size_t)m * N + n] = (_Float16)v;
        }
    }
}

// ---------------------------------------------------------------------------
// LayerNorm over 256-wide rows; one wave per row; optional elementwise add
// (positional embedding) applied AFTER norm; output as f16 (GEMM operand).
// ---------------------------------------------------------------------------
__global__ __launch_bounds__(256)
void layernorm_f16_kernel(const float* __restrict__ x,
                          const float* __restrict__ g,
                          const float* __restrict__ b,
                          const float* __restrict__ addp,
                          _Float16*    __restrict__ out,
                          int rows)
{
    const int wave = threadIdx.x >> 5;
    const int lane = threadIdx.x & 31;
    const int r = blockIdx.x * 8 + wave;
    if (r >= rows) return;

    const float* xr = x + (size_t)r * DMODEL;
    float4 a0 = ((const float4*)xr)[lane * 2 + 0];
    float4 a1 = ((const float4*)xr)[lane * 2 + 1];
    float vals[8] = {a0.x, a0.y, a0.z, a0.w, a1.x, a1.y, a1.z, a1.w};

    float s = 0.f, s2 = 0.f;
    #pragma unroll
    for (int i = 0; i < 8; ++i) { s += vals[i]; s2 += vals[i] * vals[i]; }
    s  += __shfl_xor(s, 16, 32);  s2 += __shfl_xor(s2, 16, 32);
    s  += __shfl_xor(s,  8, 32);  s2 += __shfl_xor(s2,  8, 32);
    s  += __shfl_xor(s,  4, 32);  s2 += __shfl_xor(s2,  4, 32);
    s  += __shfl_xor(s,  2, 32);  s2 += __shfl_xor(s2,  2, 32);
    s  += __shfl_xor(s,  1, 32);  s2 += __shfl_xor(s2,  1, 32);

    const float mu  = s * (1.0f / DMODEL);
    const float var = s2 * (1.0f / DMODEL) - mu * mu;
    const float rs  = rsqrtf(var + 1e-5f);

    const int c0 = lane * 8;
    #pragma unroll
    for (int i = 0; i < 8; ++i) {
        float y = (vals[i] - mu) * rs * g[c0 + i] + b[c0 + i];
        if (addp) y += addp[(size_t)r * DMODEL + c0 + i];
        out[(size_t)r * DMODEL + c0 + i] = (_Float16)y;
    }
}

// ---------------------------------------------------------------------------
// Deformable attention core + fused softmax.
// One wave per (batch*query, head): lane = head channel (HEAD_DIM == 32).
// Softmax over the 16 (level,point) logits and the 32 per-head offset floats
// live in registers and are broadcast with shfl.
// ---------------------------------------------------------------------------
__global__ __launch_bounds__(256)
void deform_attn_kernel(const float* __restrict__ value,    // [ROWS,8,32]
                        const float* __restrict__ logits,   // [ROWS,128]
                        const float* __restrict__ offsets,  // [ROWS,256]
                        const float* __restrict__ refp,     // [ROWS,4,2]
                        _Float16*    __restrict__ out16)    // [ROWS,256]
{
    const int Hs[4] = {128, 64, 32, 16};
    const int Ws[4] = {128, 64, 32, 16};
    const int St[4] = {0, 16384, 20480, 21504};

    const int lane = threadIdx.x & 31;
    const int head = threadIdx.x >> 5;   // 8 waves = 8 heads
    const int bq   = blockIdx.x;         // 0 .. ROWS-1
    const int bIdx = bq / LEN_Q;

    // ---- softmax over 16 logits (both 16-lane halves mirror each other) ----
    float lg = logits[(size_t)bq * 128 + head * 16 + (lane & 15)];
    float mx = lg;
    mx = fmaxf(mx, __shfl_xor(mx, 8, 32));
    mx = fmaxf(mx, __shfl_xor(mx, 4, 32));
    mx = fmaxf(mx, __shfl_xor(mx, 2, 32));
    mx = fmaxf(mx, __shfl_xor(mx, 1, 32));
    float e = __expf(lg - mx);
    float ssum = e;
    ssum += __shfl_xor(ssum, 8, 32);
    ssum += __shfl_xor(ssum, 4, 32);
    ssum += __shfl_xor(ssum, 2, 32);
    ssum += __shfl_xor(ssum, 1, 32);
    const float wsm = e / ssum;

    // 32 offset floats per head: [lvl][pt][xy]
    const float offv = offsets[(size_t)bq * 256 + head * 32 + lane];
    const float* rp  = refp + (size_t)bq * 8;

    // value element for global position p, this head, this lane's channel
    const float* vbase = value + (((size_t)bIdx * LEN_Q) * 8 + head) * 32 + lane;

    float acc = 0.f;
    #pragma unroll
    for (int lvl = 0; lvl < 4; ++lvl) {
        const int   Wi = Ws[lvl], Hi = Hs[lvl], start = St[lvl];
        const float Wf = (float)Wi, Hf = (float)Hi;
        const float rx = rp[lvl * 2 + 0];
        const float ry = rp[lvl * 2 + 1];
        #pragma unroll
        for (int pt = 0; pt < 4; ++pt) {
            const int ip = lvl * 4 + pt;
            const float ox = __shfl(offv, 2 * ip + 0, 32);
            const float oy = __shfl(offv, 2 * ip + 1, 32);
            const float aw = __shfl(wsm, ip, 32);

            const float x = (rx + ox / Wf) * Wf - 0.5f;
            const float y = (ry + oy / Hf) * Hf - 0.5f;
            const float x0f = floorf(x), y0f = floorf(y);
            const int   x0 = (int)x0f, y0 = (int)y0f;
            const float fx = x - x0f, fy = y - y0f;

            const float wgt[4] = {(1.f - fx) * (1.f - fy), fx * (1.f - fy),
                                  (1.f - fx) * fy,         fx * fy};
            const int xs[4] = {x0, x0 + 1, x0,     x0 + 1};
            const int ys[4] = {y0, y0,     y0 + 1, y0 + 1};

            float s = 0.f;
            #pragma unroll
            for (int c = 0; c < 4; ++c) {
                const int xc = xs[c], yc = ys[c];
                if (xc >= 0 && xc < Wi && yc >= 0 && yc < Hi) {
                    const int p = start + yc * Wi + xc;
                    s += wgt[c] * vbase[(size_t)p * 256];
                }
            }
            acc += aw * s;
        }
    }
    out16[(size_t)bq * 256 + head * 32 + lane] = (_Float16)acc;
}

// ---------------------------------------------------------------------------
// Weight prep: f32 [K,N] -> f16 transposed [N,K] (so GEMM B loads are
// contiguous in K).  And a plain f32 -> f16 converter for activations.
// ---------------------------------------------------------------------------
__global__ void convert_transpose_kernel(const float* __restrict__ w,
                                         _Float16* __restrict__ wt,
                                         int K, int N)
{
    const int t = blockIdx.x * 256 + threadIdx.x;
    if (t >= K * N) return;
    const int n = t / K, k = t % K;
    wt[t] = (_Float16)w[(size_t)k * N + n];
}

__global__ void convert_f16_kernel(const float* __restrict__ in,
                                   _Float16* __restrict__ out, int n)
{
    const int t = blockIdx.x * 256 + threadIdx.x;
    if (t < n) out[t] = (_Float16)in[t];
}

// ---------------------------------------------------------------------------
// Host launch
// ---------------------------------------------------------------------------
extern "C" void kernel_launch(void* const* d_in, const int* in_sizes, int n_in,
                              void* d_out, int out_size, void* d_ws, size_t ws_size,
                              hipStream_t stream)
{
    (void)in_sizes; (void)n_in; (void)out_size; (void)ws_size;

    const float* src    = (const float*)d_in[0];
    const float* pos    = (const float*)d_in[1];
    const float* refp   = (const float*)d_in[2];
    // d_in[3] spatial_shapes / d_in[4] level_start_index: fixed, hardcoded.
    const float* n1g    = (const float*)d_in[5];
    const float* n1b    = (const float*)d_in[6];
    const float* n2g    = (const float*)d_in[7];
    const float* n2b    = (const float*)d_in[8];
    const float* Wv     = (const float*)d_in[9];
    const float* bv     = (const float*)d_in[10];
    const float* Woff   = (const float*)d_in[11];
    const float* boff   = (const float*)d_in[12];
    const float* Wattn  = (const float*)d_in[13];
    const float* battn  = (const float*)d_in[14];
    const float* Wout   = (const float*)d_in[15];
    const float* bout   = (const float*)d_in[16];
    const float* Wf1    = (const float*)d_in[17];
    const float* bf1    = (const float*)d_in[18];
    const float* Wf2    = (const float*)d_in[19];
    const float* bf2    = (const float*)d_in[20];
    float* out          = (float*)d_out;

    // ---- carve workspace ----
    char* ws = (char*)d_ws;
    auto alloc = [&](size_t bytes) -> void* {
        void* p = (void*)ws;
        ws += (bytes + 255) & ~(size_t)255;
        return p;
    };
    const size_t RM = (size_t)ROWS * DMODEL;     // 11,141,120
    _Float16* src16   = (_Float16*)alloc(RM * 2);
    _Float16* q16     = (_Float16*)alloc(RM * 2);
    _Float16* ao16    = (_Float16*)alloc(RM * 2);           // attn core out
    _Float16* h16     = (_Float16*)alloc(RM * 2);           // LN2 out
    _Float16* f1a16   = (_Float16*)alloc((size_t)ROWS * DFFN * 2);
    float*    valbuf  = (float*)alloc(RM * 4);
    float*    offbuf  = (float*)alloc(RM * 4);
    float*    logbuf  = (float*)alloc((size_t)ROWS * 128 * 4);
    float*    src2    = (float*)alloc(RM * 4);
    _Float16* Wv16    = (_Float16*)alloc((size_t)256 * 256 * 2);
    _Float16* Woff16  = (_Float16*)alloc((size_t)256 * 256 * 2);
    _Float16* Wattn16 = (_Float16*)alloc((size_t)128 * 256 * 2);
    _Float16* Wout16  = (_Float16*)alloc((size_t)256 * 256 * 2);
    _Float16* Wf1_16  = (_Float16*)alloc((size_t)1024 * 256 * 2);
    _Float16* Wf2_16  = (_Float16*)alloc((size_t)256 * 1024 * 2);

    // ---- weight prep (tiny vs. activation traffic) ----
    convert_transpose_kernel<<<(256 * 256 + 255) / 256, 256, 0, stream>>>(Wv, Wv16, 256, 256);
    convert_transpose_kernel<<<(256 * 256 + 255) / 256, 256, 0, stream>>>(Woff, Woff16, 256, 256);
    convert_transpose_kernel<<<(256 * 128 + 255) / 256, 256, 0, stream>>>(Wattn, Wattn16, 256, 128);
    convert_transpose_kernel<<<(256 * 256 + 255) / 256, 256, 0, stream>>>(Wout, Wout16, 256, 256);
    convert_transpose_kernel<<<(256 * 1024 + 255) / 256, 256, 0, stream>>>(Wf1, Wf1_16, 256, 1024);
    convert_transpose_kernel<<<(1024 * 256 + 255) / 256, 256, 0, stream>>>(Wf2, Wf2_16, 1024, 256);

    // ---- activations ----
    convert_f16_kernel<<<(int)(RM / 256), 256, 0, stream>>>(src, src16, (int)RM);
    layernorm_f16_kernel<<<ROWS / 8, 256, 0, stream>>>(src, n1g, n1b, pos, q16, ROWS);

    // value = src @ Wv + bv                  (M,K,N = 43520,256,256)
    wmma_gemm_kernel<<<dim3(ROWS / 128, 256 / 64), 256, 0, stream>>>(
        src16, Wv16, bv, nullptr, valbuf, nullptr, ROWS, 256, 256, 0);
    // offsets = q @ Woff + boff
    wmma_gemm_kernel<<<dim3(ROWS / 128, 256 / 64), 256, 0, stream>>>(
        q16, Woff16, boff, nullptr, offbuf, nullptr, ROWS, 256, 256, 0);
    // logits = q @ Wattn + battn             (N = 128)
    wmma_gemm_kernel<<<dim3(ROWS / 128, 128 / 64), 256, 0, stream>>>(
        q16, Wattn16, battn, nullptr, logbuf, nullptr, ROWS, 256, 128, 0);

    // deformable sampling + fused softmax -> attn core output (f16)
    deform_attn_kernel<<<ROWS, 256, 0, stream>>>(valbuf, logbuf, offbuf, refp, ao16);

    // src2 = src + ao @ Wout + bout
    wmma_gemm_kernel<<<dim3(ROWS / 128, 256 / 64), 256, 0, stream>>>(
        ao16, Wout16, bout, src, src2, nullptr, ROWS, 256, 256, 0);

    // h = LN2(src2)
    layernorm_f16_kernel<<<ROWS / 8, 256, 0, stream>>>(src2, n2g, n2b, nullptr, h16, ROWS);

    // f1 = gelu(h @ Wf1 + bf1)               (N = 1024, f16 out for next GEMM)
    wmma_gemm_kernel<<<dim3(ROWS / 128, 1024 / 64), 256, 0, stream>>>(
        h16, Wf1_16, bf1, nullptr, nullptr, f1a16, ROWS, 256, 1024, 1);

    // out = src2 + f1 @ Wf2 + bf2            (K = 1024)
    wmma_gemm_kernel<<<dim3(ROWS / 128, 256 / 64), 256, 0, stream>>>(
        f1a16, Wf2_16, bf2, src2, out, nullptr, ROWS, 1024, 256, 0);
}
